// BinaryTree_65927747993695
// MI455X (gfx1250) — matrix operand order repaired and checked by reading, not compiled, not run on hardware
//
#include <hip/hip_runtime.h>
#include <hip/hip_bf16.h>

typedef __attribute__((ext_vector_type(16))) _Float16 v16h;
typedef __attribute__((ext_vector_type(8)))  float    v8f;
typedef __attribute__((ext_vector_type(4)))  float    f4v;

#define TREE_DEPTH   20
#define TREE_LEVELS  21
#define LEAF_OFFSET  ((1u << TREE_DEPTH) - 1u)
#define NDIM         128
#define WAVES_PER_BLOCK 8

// One wave32 per batch element.
//   A (16x32 f16 per K-chunk) = gathered path rows (group g -> levels g*16+m)
//   B (32x16 f16 per K-chunk) = z broadcast into all 16 columns
//   D (16x16 f32)             = logits replicated across columns
__global__ __launch_bounds__(32 * WAVES_PER_BLOCK)
void BinaryTree_65927747993695_kernel(const int* __restrict__ coll,
                                      const float* __restrict__ W,
                                      float* __restrict__ out,
                                      int nBatch)
{
    const int lane   = threadIdx.x & 31;
    const int waveId = blockIdx.x * WAVES_PER_BLOCK + (threadIdx.x >> 5);
    if (waveId >= nBatch) return;           // wave-uniform exit: EXEC stays all-1s
    const int b = waveId;

    const int hi = lane >> 4;               // which half of the wave
    const int m  = lane & 15;               // A-row owned by this lane

    const int c0 = coll[2 * b + 0];
    const int c1 = coll[2 * b + 1];
    const unsigned leaf = (unsigned)c0 + LEAF_OFFSET;       // leaf node of 'input'
    const unsigned bb   = (unsigned)c1 + LEAF_OFFSET + 1u;  // 1-based leaf id of 'context'

    const float* __restrict__ zrow = W + (size_t)leaf * NDIM;

    // ---- B fragments (z broadcast): lane<16 holds K 0..15 of chunk, lane>=16 K 16..31
    v16h bfrag[4];
#pragma unroll
    for (int c = 0; c < 4; ++c) {
        const float* zp = zrow + 32 * c + 16 * hi;
        f4v z0 = *(const f4v*)(zp + 0);
        f4v z1 = *(const f4v*)(zp + 4);
        f4v z2 = *(const f4v*)(zp + 8);
        f4v z3 = *(const f4v*)(zp + 12);
        v16h bv;
#pragma unroll
        for (int e = 0; e < 4; ++e) {
            bv[e]      = (_Float16)z0[e];
            bv[4 + e]  = (_Float16)z1[e];
            bv[8 + e]  = (_Float16)z2[e];
            bv[12 + e] = (_Float16)z3[e];
        }
        bfrag[c] = bv;
    }

    float prod = 1.0f;

#pragma unroll
    for (int g = 0; g < 2; ++g) {
        // Path node whose row this lane supplies as A-row m of the tile.
        const int lvl = g * 16 + m;
        const unsigned node = (lvl <= TREE_DEPTH) ? ((bb >> (TREE_DEPTH - lvl)) - 1u) : 0u;
        const float* __restrict__ arow = W + (size_t)node * NDIM;

        v8f acc = {};
#pragma unroll
        for (int c = 0; c < 4; ++c) {
            // A layout (matches B): lane<16 -> K 0..15; lane>=16 -> K 16..31
            const float* a0p = arow + 32 * c + 16 * hi;
            const float* a1p = arow + 32 * c + 16 * hi + 8;
            f4v x0 = *(const f4v*)(a0p + 0);
            f4v x1 = *(const f4v*)(a0p + 4);
            f4v x2 = *(const f4v*)(a1p + 0);
            f4v x3 = *(const f4v*)(a1p + 4);
            v16h av;
#pragma unroll
            for (int e = 0; e < 4; ++e) {
                av[e]      = (_Float16)x0[e];
                av[4 + e]  = (_Float16)x1[e];
                av[8 + e]  = (_Float16)x2[e];
                av[12 + e] = (_Float16)x3[e];
            }
            acc = __builtin_amdgcn_wmma_f32_16x16x32_f16(
                false, av, false, bfrag[c], (short)0, acc, false, false);
        }

        // D layout: VGPR r at this lane = logit for level g*16 + 8*hi + r
#pragma unroll
        for (int r = 0; r < 8; ++r) {
            const int dl = g * 16 + 8 * hi + r;
            const float s = 1.0f / (1.0f + __expf(-acc[r]));
            prod *= (dl < TREE_LEVELS) ? s : 1.0f;
        }
    }

    // lanes 0..15 hold levels {0..7,16..20}; lanes 16..31 hold {8..15}
    prod *= __shfl_xor(prod, 16, 32);
    if (lane == 0) out[b] = prod;
}

extern "C" void kernel_launch(void* const* d_in, const int* in_sizes, int n_in,
                              void* d_out, int out_size, void* d_ws, size_t ws_size,
                              hipStream_t stream) {
    const int*   coll = (const int*)d_in[0];     // [B,2] int32
    const float* W    = (const float*)d_in[1];   // [SIZE, 128] f32
    float*       out  = (float*)d_out;           // [B] f32
    const int nBatch  = in_sizes[0] / 2;
    const int blocks  = (nBatch + WAVES_PER_BLOCK - 1) / WAVES_PER_BLOCK;
    BinaryTree_65927747993695_kernel<<<blocks, 32 * WAVES_PER_BLOCK, 0, stream>>>(
        coll, W, out, nBatch);
}
